// EnsembleRSSM_83760452206942
// MI455X (gfx1250) — matrix operand order repaired and checked by reading, not compiled, not run on hardware
//
#include <hip/hip_runtime.h>
#include <hip/hip_bf16.h>

// ---------------- problem dims ----------------
#define Bc    256
#define Tc    64
#define EMBc  1536
#define ACTc  18
#define Sc    32
#define Dc    32
#define Mc    600
#define MPc   608      // M padded to 16
#define SDc   1024     // S*D
#define NGc   1824     // 3*608 concatenated GRU gates
#define KT1   33       // K-tiles (32 wide) for W1 (1024+18 -> 1056)
#define KTM   19       // K-tiles for M=608
#define KT2   67       // K-tiles for W2 (600+1536 -> 2144)
#define MTILES 16      // B/16
#define WAVES 4
#define KCHUNK 32      // A-stripe LDS staging chunk (32 tiles = 32KB)

typedef __attribute__((ext_vector_type(16))) __bf16 v16bf;
typedef __attribute__((ext_vector_type(8)))  float  v8f;

// ---------------- RNG helpers ----------------
__device__ __forceinline__ unsigned hash_u32(unsigned x) {
  x ^= x >> 16; x *= 0x7feb352du; x ^= x >> 15; x *= 0x846ca68bu; x ^= x >> 16;
  return x;
}
__device__ __forceinline__ float u01(unsigned x) {
  return ((hash_u32(x) >> 8) + 0.5f) * (1.0f / 16777216.0f);
}

// ---------------- WMMA GEMM with LDS-staged A ----------------
// A_pk: fragment-packed bf16 A tiles [Mtiles][Kt][512]; per tile, lane-major 16 bf16
// B_pk: fragment-packed bf16 B tiles [Ntiles][Kt][512]
// All 4 waves of a block share one M-tile; the A stripe is staged through LDS in
// 32-KB chunks and consumed via ds loads, B streams from L2 via b128 loads.
// C: f32 row-major [256][ldC]; act: 0=none, 1=elu
__global__ __launch_bounds__(128) void gemm_bf16(
    const __bf16* __restrict__ Apk, const __bf16* __restrict__ Bpk,
    const float* __restrict__ bias, float* __restrict__ C,
    int Kt, int Ntiles, int ldC, int act)
{
  __shared__ __align__(16) __bf16 As[KCHUNK * 512];  // 32 KB

  const int lane = threadIdx.x & 31;
  const int wave = threadIdx.x >> 5;
  const int mt = blockIdx.x;
  const int nt = blockIdx.y * WAVES + wave;
  const bool active = (nt < Ntiles);

  const v16bf* Bf = ((const v16bf*)Bpk) + (size_t)(active ? nt : 0) * Kt * 32 + lane;

  v8f acc = {};
  for (int kc = 0; kc < Kt; kc += KCHUNK) {
    const int kn = (Kt - kc < KCHUNK) ? (Kt - kc) : KCHUNK;
    __syncthreads();
    // cooperative global -> LDS copy of kn A tiles (kn*1024 bytes)
    {
      const uint4* Ag = (const uint4*)(((const v16bf*)Apk) + ((size_t)mt * Kt + kc) * 32);
      uint4* Asv = (uint4*)As;
      for (int i = threadIdx.x; i < kn * 64; i += 128) Asv[i] = Ag[i];
    }
    __syncthreads();
    if (active) {
      if (kc + kn < Kt)
        __builtin_prefetch((const void*)(Bf + (size_t)(kc + kn) * 32), 0, 3);
      const v16bf* Afs = ((const v16bf*)As) + lane;
      const v16bf* Bcc = Bf + (size_t)kc * 32;
      for (int kt = 0; kt < kn; ++kt) {
        v16bf a = Afs[kt * 32];   // ds_load_b128 x2
        v16bf b = Bcc[kt * 32];   // global_load_b128 x2
        acc = __builtin_amdgcn_wmma_f32_16x16x32_bf16(
            false, a, false, b, (short)0, acc, false, false);
      }
    }
  }

  if (active) {
    const int n = nt * 16 + (lane & 15);
    const float bv = bias ? bias[n] : 0.0f;
    const int mbase = mt * 16 + ((lane >> 4) << 3);  // M = v + 8*(lane/16)
#pragma unroll
    for (int v = 0; v < 8; ++v) {
      float val = acc[v] + bv;
      if (act == 1) val = val > 0.0f ? val : (__expf(val) - 1.0f);
      C[(size_t)(mbase + v) * ldC + n] = val;
    }
  }
}

// ---------------- activation packing (f32 -> bf16 A-fragment layout) ----------------
// k < w0 from s0, k < w0+w1 from s1, else 0.  optional row mask (1 - is_first[b,t]).
__global__ void pack_a(__bf16* __restrict__ dst,
                       const float* __restrict__ s0, int ld0, int w0,
                       const float* __restrict__ s1, int ld1, int w1,
                       int Kt, const float* __restrict__ isf, int t, int total)
{
  int idx = blockIdx.x * blockDim.x + threadIdx.x;
  if (idx >= total) return;
  const int e = idx & 15;
  const int lane = (idx >> 4) & 31;
  const int tile = idx >> 9;
  const int kt = tile % Kt;
  const int mt = tile / Kt;
  const int m = lane & 15, half = lane >> 4;
  const int v = e >> 1;
  const int kp = (v < 4 ? 2 * v : 16 + 2 * (v - 4)) + half * 8;
  const int k = kt * 32 + kp + (e & 1);
  const int row = mt * 16 + m;
  float val = 0.0f;
  if (k < w0) val = s0[(size_t)row * ld0 + k];
  else if (s1 && k < w0 + w1) val = s1[(size_t)row * ld1 + (k - w0)];
  if (isf) val *= (1.0f - isf[row * Tc + t]);
  dst[idx] = (__bf16)val;
}

// ---------------- weight packing (f32 row-major [K][N] -> bf16 B-fragment layout) ----
__global__ void pack_b(__bf16* __restrict__ dst, const float* __restrict__ W,
                       int K, int N, int Kt, int total)
{
  int idx = blockIdx.x * blockDim.x + threadIdx.x;
  if (idx >= total) return;
  const int e = idx & 15;
  const int lane = (idx >> 4) & 31;
  const int tile = idx >> 9;
  const int kt = tile % Kt;
  const int nt = tile / Kt;
  const int nl = lane & 15, hg = lane >> 4, v = e >> 1;
  const int k = kt * 32 + (v < 4 ? 2 * v : 16 + 2 * (v - 4)) + hg * 8 + (e & 1);
  const int n = nt * 16 + nl;
  float val = (k < K && n < N) ? W[(size_t)k * N + n] : 0.0f;
  dst[idx] = (__bf16)val;
}

// three 600x600 weights concatenated along N into segments of 608 (total N=1824)
__global__ void pack_b3(__bf16* __restrict__ dst,
                        const float* __restrict__ W0, const float* __restrict__ W1,
                        const float* __restrict__ W2, int Kt, int total)
{
  int idx = blockIdx.x * blockDim.x + threadIdx.x;
  if (idx >= total) return;
  const int e = idx & 15;
  const int lane = (idx >> 4) & 31;
  const int tile = idx >> 9;
  const int kt = tile % Kt;
  const int nt = tile / Kt;
  const int nl = lane & 15, hg = lane >> 4, v = e >> 1;
  const int k = kt * 32 + (v < 4 ? 2 * v : 16 + 2 * (v - 4)) + hg * 8 + (e & 1);
  const int n = nt * 16 + nl;
  const int seg = n / MPc, nn = n % MPc;
  const float* W = seg == 0 ? W0 : (seg == 1 ? W1 : W2);
  float val = (k < Mc && nn < Mc) ? W[(size_t)k * Mc + nn] : 0.0f;
  dst[idx] = (__bf16)val;
}

// ---------------- bias padding ----------------
__global__ void bias_pad(float* __restrict__ dst, const float* __restrict__ s0,
                         const float* __restrict__ s1, const float* __restrict__ s2,
                         int realN, int padN, int nseg)
{
  int idx = blockIdx.x * blockDim.x + threadIdx.x;
  if (idx >= nseg * padN) return;
  const int seg = idx / padN, nn = idx % padN;
  const float* s = seg == 0 ? s0 : (seg == 1 ? s1 : s2);
  dst[idx] = (nn < realN) ? s[nn] : 0.0f;
}

// ---------------- GRU combine ----------------
__global__ void gru_combine(const float* __restrict__ dcat, const float* __restrict__ xcat,
                            const float* __restrict__ x, float* __restrict__ det_state,
                            float* __restrict__ outDet, int t)
{
  int idx = blockIdx.x * blockDim.x + threadIdx.x;
  if (idx >= Bc * MPc) return;
  const int b = idx / MPc, n = idx % MPc;
  const size_t g = (size_t)b * NGc;
  const float dr = dcat[g + n],           xr = xcat[g + n];
  const float dz = dcat[g + MPc + n],     xz = xcat[g + MPc + n];
  const float dn = dcat[g + 2 * MPc + n], xn = xcat[g + 2 * MPc + n];
  const float r = 1.0f / (1.0f + __expf(-(dr + xr)));
  const float z = 1.0f / (1.0f + __expf(-(dz + xz)));
  const float nn2 = tanhf(dn + r * xn);
  float nd = (1.0f - z) * nn2 + z * x[idx];
  if (n >= Mc) nd = 0.0f;
  det_state[idx] = nd;
  if (outDet && n < Mc) outDet[((size_t)b * Tc + t) * Mc + n] = nd;
}

// ---------------- categorical gumbel-max sample + stats copy ----------------
__global__ void sample_k(const float* __restrict__ logits, float* __restrict__ statsOut,
                         float* __restrict__ stochOut, float* __restrict__ state,
                         int t, unsigned tag)
{
  int idx = blockIdx.x * blockDim.x + threadIdx.x;
  if (idx >= Bc * Sc) return;
  const int b = idx / Sc, s = idx % Sc;
  const float* lg = logits + (size_t)b * SDc + s * Dc;
  const size_t obase = ((size_t)b * Tc + t) * SDc + s * Dc;
  float best = -1e30f; int bi = 0;
  for (int d = 0; d < Dc; ++d) {
    const float l = lg[d];
    statsOut[obase + d] = l;
    const unsigned h = hash_u32(tag ^ hash_u32((unsigned)idx * 977u + (unsigned)(t * Dc + d)));
    const float g = l - __logf(-__logf(u01(h)));
    if (g > best) { best = g; bi = d; }
  }
  for (int d = 0; d < Dc; ++d) {
    const float oh = (d == bi) ? 1.0f : 0.0f;
    stochOut[obase + d] = oh;
    if (state) state[(size_t)b * SDc + s * Dc + d] = oh;
  }
}

// ---------------- KL(post || prior) per batch row ----------------
__global__ void kl_k(const float* __restrict__ lq, const float* __restrict__ lp,
                     float* __restrict__ klv, int t)
{
  int b = blockIdx.x * blockDim.x + threadIdx.x;
  if (b >= Bc) return;
  float total = 0.0f;
  for (int s = 0; s < Sc; ++s) {
    const float* q = lq + (size_t)b * SDc + s * Dc;
    const float* p = lp + (size_t)b * SDc + s * Dc;
    float mq = -1e30f, mp_ = -1e30f;
    for (int d = 0; d < Dc; ++d) { mq = fmaxf(mq, q[d]); mp_ = fmaxf(mp_, p[d]); }
    float sq = 0.0f, sp = 0.0f;
    for (int d = 0; d < Dc; ++d) { sq += __expf(q[d] - mq); sp += __expf(p[d] - mp_); }
    const float lzq = mq + __logf(sq), lzp = mp_ + __logf(sp);
    for (int d = 0; d < Dc; ++d) {
      const float ql = q[d] - lzq;
      total += __expf(ql) * (ql - (p[d] - lzp));
    }
  }
  klv[b * Tc + t] = total;
}

__global__ void kl_reduce(const float* __restrict__ klv, float* __restrict__ kl_loss)
{
  __shared__ float sm[256];
  float acc = 0.0f;
  for (int i = threadIdx.x; i < Bc * Tc; i += 256) acc += klv[i];
  sm[threadIdx.x] = acc;
  __syncthreads();
  for (int off = 128; off > 0; off >>= 1) {
    if (threadIdx.x < off) sm[threadIdx.x] += sm[threadIdx.x + off];
    __syncthreads();
  }
  if (threadIdx.x == 0) {
    const float m = sm[0] / (float)(Bc * Tc);
    *kl_loss = m > 1.0f ? m : 1.0f;
  }
}

// ---------------- state init (stoch0 uniform-hash, det0 zero) ----------------
__global__ void init_state(float* __restrict__ stoch_state, float* __restrict__ det_state)
{
  int idx = blockIdx.x * blockDim.x + threadIdx.x;
  if (idx < Bc * SDc) stoch_state[idx] = u01(0x9E3779B9u ^ (unsigned)idx);
  if (idx < Bc * MPc) det_state[idx] = 0.0f;
}

// =========================== host orchestration ===========================
#define GRID1(n) dim3(((n) + 255) / 256), dim3(256)

extern "C" void kernel_launch(void* const* d_in, const int* in_sizes, int n_in,
                              void* d_out, int out_size, void* d_ws, size_t ws_size,
                              hipStream_t stream) {
  (void)in_sizes; (void)n_in; (void)out_size; (void)ws_size;
  const float* embed  = (const float*)d_in[0];
  const float* action = (const float*)d_in[1];
  const float* isf    = (const float*)d_in[2];
  const float* W1  = (const float*)d_in[3];  const float* b1  = (const float*)d_in[4];
  const float* Wir = (const float*)d_in[5];  const float* bir = (const float*)d_in[6];
  const float* Wiz = (const float*)d_in[7];  const float* biz = (const float*)d_in[8];
  const float* Win = (const float*)d_in[9];  const float* bin_ = (const float*)d_in[10];
  const float* Whr = (const float*)d_in[11]; const float* Whz = (const float*)d_in[12];
  const float* Whn = (const float*)d_in[13];
  const float* We1 = (const float*)d_in[14]; const float* be1 = (const float*)d_in[15];
  const float* We2 = (const float*)d_in[16]; const float* be2 = (const float*)d_in[17];
  const float* W2  = (const float*)d_in[18]; const float* b2  = (const float*)d_in[19];
  const float* W3  = (const float*)d_in[20]; const float* b3  = (const float*)d_in[21];
  float* out = (float*)d_out;

  // output offsets (reference tuple order, flat concat)
  const size_t o_post_stats  = 0;
  const size_t o_post_stoch  = o_post_stats + (size_t)Bc * Tc * SDc;
  const size_t o_deter       = o_post_stoch + (size_t)Bc * Tc * SDc;
  const size_t o_prior_stats = o_deter + (size_t)Bc * Tc * Mc;
  const size_t o_prior_stoch = o_prior_stats + (size_t)Bc * Tc * SDc;
  const size_t o_kl_loss     = o_prior_stoch + (size_t)Bc * Tc * SDc;
  const size_t o_kl_value    = o_kl_loss + 1;
  const size_t o_img_stats   = o_kl_value + (size_t)Bc * Tc;
  const size_t o_img_stoch   = o_img_stats + (size_t)Bc * Tc * SDc;
  const size_t o_img_det     = o_img_stoch + (size_t)Bc * Tc * SDc;

  // workspace carve
  char* ws = (char*)d_ws;
  auto alloc_bf = [&](size_t n) { __bf16* p = (__bf16*)ws; ws += ((n * 2 + 255) / 256) * 256; return p; };
  auto alloc_f  = [&](size_t n) { float*  p = (float*)ws;  ws += ((n * 4 + 255) / 256) * 256; return p; };

  __bf16* W1_pk  = alloc_bf((size_t)KT1 * 38 * 512);
  __bf16* Wi_pk  = alloc_bf((size_t)KTM * 114 * 512);
  __bf16* Wh_pk  = alloc_bf((size_t)KTM * 114 * 512);
  __bf16* We1_pk = alloc_bf((size_t)KTM * 38 * 512);
  __bf16* We2_pk = alloc_bf((size_t)KTM * 64 * 512);
  __bf16* W2_pk  = alloc_bf((size_t)KT2 * 38 * 512);
  __bf16* W3_pk  = alloc_bf((size_t)KTM * 64 * 512);
  __bf16* in1_pk  = alloc_bf((size_t)MTILES * KT1 * 512);
  __bf16* det_pk  = alloc_bf((size_t)MTILES * KTM * 512);
  __bf16* x_pk    = alloc_bf((size_t)MTILES * KTM * 512);
  __bf16* ndet_pk = alloc_bf((size_t)MTILES * KTM * 512);
  __bf16* h_pk    = alloc_bf((size_t)MTILES * KTM * 512);
  __bf16* cat2_pk = alloc_bf((size_t)MTILES * KT2 * 512);
  __bf16* xp_pk   = alloc_bf((size_t)MTILES * KTM * 512);

  float* b1p  = alloc_f(MPc);
  float* bgru = alloc_f(NGc);
  float* be1p = alloc_f(MPc);
  float* be2p = alloc_f(SDc);
  float* b2p  = alloc_f(MPc);
  float* b3p  = alloc_f(SDc);
  float* stoch_state = alloc_f((size_t)Bc * SDc);
  float* det_state   = alloc_f((size_t)Bc * MPc);
  float* x_f32  = alloc_f((size_t)Bc * MPc);
  float* dcat   = alloc_f((size_t)Bc * NGc);
  float* xcat   = alloc_f((size_t)Bc * NGc);
  float* h_f32  = alloc_f((size_t)Bc * MPc);
  float* lp_f32 = alloc_f((size_t)Bc * SDc);
  float* xp_f32 = alloc_f((size_t)Bc * MPc);
  float* lq_f32 = alloc_f((size_t)Bc * SDc);

  // ---- one-time weight/bias packing ----
  {
    int n;
    n = KT1 * 38 * 512;  pack_b<<<GRID1(n), 0, stream>>>(W1_pk,  W1,  SDc + ACTc, Mc,  KT1, n);
    n = KTM * 114 * 512; pack_b3<<<GRID1(n), 0, stream>>>(Wi_pk, Wir, Wiz, Win, KTM, n);
    n = KTM * 114 * 512; pack_b3<<<GRID1(n), 0, stream>>>(Wh_pk, Whr, Whz, Whn, KTM, n);
    n = KTM * 38 * 512;  pack_b<<<GRID1(n), 0, stream>>>(We1_pk, We1, Mc, Mc,  KTM, n);
    n = KTM * 64 * 512;  pack_b<<<GRID1(n), 0, stream>>>(We2_pk, We2, Mc, SDc, KTM, n);
    n = KT2 * 38 * 512;  pack_b<<<GRID1(n), 0, stream>>>(W2_pk,  W2,  Mc + EMBc, Mc, KT2, n);
    n = KTM * 64 * 512;  pack_b<<<GRID1(n), 0, stream>>>(W3_pk,  W3,  Mc, SDc, KTM, n);
    bias_pad<<<GRID1(MPc), 0, stream>>>(b1p,  b1,  b1,  b1,  Mc,  MPc, 1);
    bias_pad<<<GRID1(NGc), 0, stream>>>(bgru, bir, biz, bin_, Mc, MPc, 3);
    bias_pad<<<GRID1(MPc), 0, stream>>>(be1p, be1, be1, be1, Mc,  MPc, 1);
    bias_pad<<<GRID1(SDc), 0, stream>>>(be2p, be2, be2, be2, SDc, SDc, 1);
    bias_pad<<<GRID1(MPc), 0, stream>>>(b2p,  b2,  b2,  b2,  Mc,  MPc, 1);
    bias_pad<<<GRID1(SDc), 0, stream>>>(b3p,  b3,  b3,  b3,  SDc, SDc, 1);
  }

  const int tot_in1  = MTILES * KT1 * 512;
  const int tot_m    = MTILES * KTM * 512;
  const int tot_cat2 = MTILES * KT2 * 512;
  const dim3 blk(128);
  const dim3 g38(MTILES, (38 + WAVES - 1) / WAVES);
  const dim3 g64(MTILES, (64 + WAVES - 1) / WAVES);
  const dim3 g114(MTILES, (114 + WAVES - 1) / WAVES);

  // shared img_core: pack inputs, W1 GEMM, GRU, prior head
  auto img_core = [&](int t, const float* maskp, float* detOut) {
    pack_a<<<GRID1(tot_in1), 0, stream>>>(in1_pk, stoch_state, SDc, SDc,
                                          action + (size_t)t * ACTc, Tc * ACTc, ACTc,
                                          KT1, maskp, t, tot_in1);
    pack_a<<<GRID1(tot_m), 0, stream>>>(det_pk, det_state, MPc, MPc,
                                        nullptr, 0, 0, KTM, maskp, t, tot_m);
    gemm_bf16<<<g38, blk, 0, stream>>>(in1_pk, W1_pk, b1p, x_f32, KT1, 38, MPc, 1);
    pack_a<<<GRID1(tot_m), 0, stream>>>(x_pk, x_f32, MPc, MPc, nullptr, 0, 0,
                                        KTM, nullptr, 0, tot_m);
    gemm_bf16<<<g114, blk, 0, stream>>>(det_pk, Wi_pk, bgru, dcat, KTM, 114, NGc, 0);
    gemm_bf16<<<g114, blk, 0, stream>>>(x_pk,   Wh_pk, nullptr, xcat, KTM, 114, NGc, 0);
    gru_combine<<<GRID1(Bc * MPc), 0, stream>>>(dcat, xcat, x_f32, det_state, detOut, t);
    pack_a<<<GRID1(tot_m), 0, stream>>>(ndet_pk, det_state, MPc, MPc, nullptr, 0, 0,
                                        KTM, nullptr, 0, tot_m);
    gemm_bf16<<<g38, blk, 0, stream>>>(ndet_pk, We1_pk, be1p, h_f32, KTM, 38, MPc, 1);
    pack_a<<<GRID1(tot_m), 0, stream>>>(h_pk, h_f32, MPc, MPc, nullptr, 0, 0,
                                        KTM, nullptr, 0, tot_m);
    gemm_bf16<<<g64, blk, 0, stream>>>(h_pk, We2_pk, be2p, lp_f32, KTM, 64, SDc, 0);
  };

  // ---- observe scan ----
  init_state<<<GRID1(Bc * SDc), 0, stream>>>(stoch_state, det_state);
  for (int t = 0; t < Tc; ++t) {
    img_core(t, isf, out + o_deter);
    sample_k<<<GRID1(Bc * Sc), 0, stream>>>(lp_f32, out + o_prior_stats,
                                            out + o_prior_stoch, nullptr, t, 0xA0000001u);
    pack_a<<<GRID1(tot_cat2), 0, stream>>>(cat2_pk, det_state, MPc, Mc,
                                           embed + (size_t)t * EMBc, Tc * EMBc, EMBc,
                                           KT2, nullptr, 0, tot_cat2);
    gemm_bf16<<<g38, blk, 0, stream>>>(cat2_pk, W2_pk, b2p, xp_f32, KT2, 38, MPc, 1);
    pack_a<<<GRID1(tot_m), 0, stream>>>(xp_pk, xp_f32, MPc, MPc, nullptr, 0, 0,
                                        KTM, nullptr, 0, tot_m);
    gemm_bf16<<<g64, blk, 0, stream>>>(xp_pk, W3_pk, b3p, lq_f32, KTM, 64, SDc, 0);
    sample_k<<<GRID1(Bc * Sc), 0, stream>>>(lq_f32, out + o_post_stats,
                                            out + o_post_stoch, stoch_state, t, 0xB0000001u);
    kl_k<<<GRID1(Bc), 0, stream>>>(lq_f32, lp_f32, out + o_kl_value, t);
  }
  kl_reduce<<<dim3(1), dim3(256), 0, stream>>>(out + o_kl_value, out + o_kl_loss);

  // ---- imagine rollout (open loop from initial state) ----
  init_state<<<GRID1(Bc * SDc), 0, stream>>>(stoch_state, det_state);
  for (int t = 0; t < Tc; ++t) {
    img_core(t, nullptr, out + o_img_det);
    sample_k<<<GRID1(Bc * Sc), 0, stream>>>(lp_f32, out + o_img_stats,
                                            out + o_img_stoch, stoch_state, t, 0xC0000001u);
  }
}
